// DecomposedMultiheadAttention_72705206386707
// MI455X (gfx1250) — matrix operand order repaired and checked by reading, compiled, not run on hardware
//
#include <hip/hip_runtime.h>
#include <hip/hip_bf16.h>

// ---------------------------------------------------------------------------
// Problem constants (match reference): B=2, S=2048, E=1024, H=16, D=64
// ---------------------------------------------------------------------------
#define BB 2
#define SS 2048
#define EE 1024
#define HH 16
#define DD 64
#define MTOT (BB * SS)   // 4096 rows for the big GEMMs
#define KDIM EE          // 1024
#define NDIM EE          // 1024

typedef _Float16 half_t;
typedef __attribute__((ext_vector_type(16))) _Float16 v16h;
typedef __attribute__((ext_vector_type(8)))  _Float16 v8h;
typedef __attribute__((ext_vector_type(4)))  _Float16 v4h;
typedef __attribute__((ext_vector_type(8)))  float    v8f;

static __device__ __forceinline__ v16h hcat(v8h lo, v8h hi) {
    return __builtin_shufflevector(lo, hi, 0, 1, 2, 3, 4, 5, 6, 7,
                                           8, 9, 10, 11, 12, 13, 14, 15);
}
static __device__ __forceinline__ v8f wmma_f16(v16h a, v16h b, v8f c) {
    // D = A(16x32 f16) * B(32x16 f16) + C(16x16 f32)
    return __builtin_amdgcn_wmma_f32_16x16x32_f16(false, a, false, b,
                                                  (short)0, c, false, false);
}
static __device__ __forceinline__ v8f vzero8() {
    v8f z;
#pragma unroll
    for (int i = 0; i < 8; ++i) z[i] = 0.0f;
    return z;
}

// Async copy of 32 bytes (2 x b128, INST_OFFSET applies to both global and LDS
// sides per the CDNA5 async pseudocode) from global to LDS. Tracked by ASYNCcnt.
static __device__ __forceinline__ void async_cp32(void* lds, const half_t* gsrc) {
    unsigned l = (unsigned)(size_t)lds;  // LDS byte address for VDST operand
    asm volatile("global_load_async_to_lds_b128 %0, %1, off"
                 :: "v"(l), "v"(gsrc) : "memory");
    asm volatile("global_load_async_to_lds_b128 %0, %1, off offset:16"
                 :: "v"(l), "v"(gsrc) : "memory");
}
static __device__ __forceinline__ void wait_async_le4() {
    asm volatile("s_wait_asynccnt 0x4" ::: "memory");
}
static __device__ __forceinline__ void wait_async_0() {
    asm volatile("s_wait_asynccnt 0x0" ::: "memory");
}

// ---------------------------------------------------------------------------
// Kernel 1: f32 -> f16 conversion (vectorized x4)
// ---------------------------------------------------------------------------
__global__ __launch_bounds__(256) void cvt_f32_f16(const float* __restrict__ in,
                                                   half_t* __restrict__ out,
                                                   int n4) {
    int i = blockIdx.x * blockDim.x + threadIdx.x;
    if (i < n4) {
        float4 v = ((const float4*)in)[i];
        v4h h;
        h[0] = (half_t)v.x; h[1] = (half_t)v.y;
        h[2] = (half_t)v.z; h[3] = (half_t)v.w;
        ((v4h*)out)[i] = h;
    }
}

// ---------------------------------------------------------------------------
// Kernel 2: WMMA GEMM  C[M,N] = A[M,K] * Bt[N,K]^T + bias
//   mode 0: store f16 to [B,H,S,D]   (Q/K projections, per-head layout)
//   mode 1: store f16 to [B,H,D,S]   (V projection, transposed for P*V)
//   mode 2: store f32 to [M,N]       (output projection)
// Block tile 128x128, BK=64, 8 waves each computing 32x64 via 2x4 WMMA tiles.
// Double-buffered LDS filled by async-to-LDS copies (software pipeline).
// ---------------------------------------------------------------------------
__global__ __launch_bounds__(256) void gemm_wmma(const half_t* __restrict__ A,
                                                 const half_t* __restrict__ Bt,
                                                 const float*  __restrict__ bias,
                                                 void* __restrict__ Cout,
                                                 int mode) {
    __shared__ __align__(16) half_t As[2][128][72];   // 2 x (128 x 64 (+8 pad))
    __shared__ __align__(16) half_t Bs[2][128][72];

    const int tid  = threadIdx.x;
    const int lane = tid & 31;
    const int wave = tid >> 5;
    const int lh   = lane & 15;
    const int hi   = (lane >> 4) & 1;

    const int ntile = blockIdx.x & 7;        // N/128 = 8
    const int mtile = blockIdx.x >> 3;       // M/128 = 32
    const int m0 = mtile * 128, n0 = ntile * 128;
    const int wm0 = (wave >> 1) * 32;        // 4 waves along M
    const int wn0 = (wave & 1) * 64;         // 2 waves along N

    v8f acc[2][4];
#pragma unroll
    for (int mi = 0; mi < 2; ++mi)
#pragma unroll
        for (int ni = 0; ni < 4; ++ni) acc[mi][ni] = vzero8();

    const int sr = tid >> 1;                 // staging row 0..127
    const int scol = (tid & 1) * 32;         // staging col offset (halves)

    // Issue async copies for the k-block starting at `kb` into buffer `buf`
    // (4 async b128 ops per thread per call).
    auto issue = [&](int kb, int buf) {
        async_cp32(&As[buf][sr][scol], A  + (size_t)(m0 + sr) * KDIM + kb + scol);
        async_cp32(&Bs[buf][sr][scol], Bt + (size_t)(n0 + sr) * KDIM + kb + scol);
    };

    issue(0, 0);  // prologue: stage first tile
    for (int kb = 0, it = 0; kb < KDIM; kb += 64, ++it) {
        const int cur = it & 1;
        const bool more = (kb + 64 < KDIM);
        if (more) {
            issue(kb + 64, cur ^ 1);   // stream next tile while computing this one
            wait_async_le4();          // current tile's 4 ops have completed
        } else {
            wait_async_0();
        }
        __syncthreads();

#pragma unroll
        for (int kk = 0; kk < 64; kk += 32) {
            v16h af[2], bf[4];
#pragma unroll
            for (int mi = 0; mi < 2; ++mi) {
                // A frag: lane<16 holds K {0..7,16..23}, lane>=16 {8..15,24..31}
                const half_t* ap = &As[cur][wm0 + mi * 16 + lh][kk + hi * 8];
                af[mi] = hcat(*(const v8h*)ap, *(const v8h*)(ap + 16));
            }
#pragma unroll
            for (int ni = 0; ni < 4; ++ni) {
                // B frag: column n of B = row n of Bt; lane<16 K {0..7,16..23}, lane>=16 {8..15,24..31}
                const half_t* bp = &Bs[cur][wn0 + ni * 16 + lh][kk + hi * 8];
                bf[ni] = hcat(*(const v8h*)bp, *(const v8h*)(bp + 16));
            }
#pragma unroll
            for (int mi = 0; mi < 2; ++mi)
#pragma unroll
                for (int ni = 0; ni < 4; ++ni)
                    acc[mi][ni] = wmma_f16(af[mi], bf[ni], acc[mi][ni]);
        }
        __syncthreads();  // all reads of buf[cur] done before it is refilled
    }

    // ---- epilogue ----
#pragma unroll
    for (int mi = 0; mi < 2; ++mi) {
#pragma unroll
        for (int ni = 0; ni < 4; ++ni) {
            const int cmBase = m0 + wm0 + mi * 16 + hi * 8;
            const int cn = n0 + wn0 + ni * 16 + lh;
            const float bval = bias[cn];
#pragma unroll
            for (int r = 0; r < 8; ++r) {
                const int mm = cmBase + r;
                const float val = acc[mi][ni][r] + bval;
                if (mode == 2) {
                    ((float*)Cout)[(size_t)mm * NDIM + cn] = val;
                } else {
                    const int bb = mm >> 11, s = mm & (SS - 1);   // mm = b*S + s
                    const int h = cn >> 6,  d = cn & (DD - 1);    // cn = h*D + d
                    const half_t hv = (half_t)val;
                    if (mode == 0)  // [B,H,S,D]
                        ((half_t*)Cout)[(((size_t)(bb * HH + h)) * SS + s) * DD + d] = hv;
                    else            // [B,H,D,S]
                        ((half_t*)Cout)[(((size_t)(bb * HH + h)) * DD + d) * SS + s] = hv;
                }
            }
        }
    }
}

// ---------------------------------------------------------------------------
// Kernel 3: streaming (flash-style) attention.
// Grid: 512 blocks = (B*H=32) x (S/128=16 query chunks); 256 threads = 8 waves.
// Each wave owns 16 query rows. K/V tiles of 128 keys streamed through
// double-buffered LDS via async-to-LDS copies.
// ---------------------------------------------------------------------------
__global__ __launch_bounds__(256) void attn_wmma(const half_t* __restrict__ qh,  // [B,H,S,D]
                                                 const half_t* __restrict__ kh,  // [B,H,S,D]
                                                 const half_t* __restrict__ vt,  // [B,H,D,S]
                                                 half_t* __restrict__ ao) {      // [B,S,E]
    __shared__ __align__(16) half_t Ks[2][128][72];       // keys x d
    __shared__ __align__(16) half_t Vs[2][64][136];       // d x keys
    __shared__ __align__(16) half_t Ps[8][16][136];       // per-wave probs

    const int tid  = threadIdx.x;
    const int lane = tid & 31;
    const int wave = tid >> 5;
    const int lh   = lane & 15;
    const int hi   = (lane >> 4) & 1;

    const int qc = blockIdx.x & 15;   // query chunk
    const int bh = blockIdx.x >> 4;   // b*H + h
    const half_t* qb = qh + (size_t)bh * SS * DD;
    const half_t* kb = kh + (size_t)bh * SS * DD;
    const half_t* vb = vt + (size_t)bh * DD * SS;
    const int q0 = qc * 128 + wave * 16;

    // Q fragments for this wave's 16 rows (kept in registers for whole kernel)
    v16h aq[2];
#pragma unroll
    for (int kki = 0; kki < 2; ++kki) {
        const half_t* qp = qb + (size_t)(q0 + lh) * DD + kki * 32 + hi * 8;
        aq[kki] = hcat(*(const v8h*)qp, *(const v8h*)(qp + 16));
    }

    float rmax[8], rsum[8];
    v8f o[4];
#pragma unroll
    for (int r = 0; r < 8; ++r) { rmax[r] = -1e30f; rsum[r] = 0.0f; }
#pragma unroll
    for (int n = 0; n < 4; ++n) o[n] = vzero8();

    const int sr = tid >> 1, scol = (tid & 1) * 32;   // K staging: 128 x 64
    const int vr = tid >> 2, vcol = (tid & 3) * 32;   // V staging: 64 x 128

    auto issue = [&](int kt, int buf) {
        async_cp32(&Ks[buf][sr][scol], kb + (size_t)(kt * 128 + sr) * DD + scol);
        async_cp32(&Vs[buf][vr][vcol], vb + (size_t)vr * SS + kt * 128 + vcol);
    };

    const int NT = SS / 128;  // 16 key tiles
    issue(0, 0);
    for (int kt = 0; kt < NT; ++kt) {
        const int cur = kt & 1;
        const bool more = (kt + 1 < NT);
        if (more) {
            issue(kt + 1, cur ^ 1);
            wait_async_le4();
        } else {
            wait_async_0();
        }
        __syncthreads();

        // ---- scores: 16 queries x 128 keys (8 WMMA tiles, K-dim = D = 64) ----
        v8f sca[8];
#pragma unroll
        for (int t = 0; t < 8; ++t) sca[t] = vzero8();
#pragma unroll
        for (int kki = 0; kki < 2; ++kki) {
#pragma unroll
            for (int t = 0; t < 8; ++t) {
                const half_t* kp = &Ks[cur][t * 16 + lh][kki * 32 + hi * 8];
                v16h bf = hcat(*(const v8h*)kp, *(const v8h*)(kp + 16));
                sca[t] = wmma_f16(aq[kki], bf, sca[t]);
            }
        }

        // ---- online softmax (C-layout: row = r + hi*8, key = t*16 + lane%16) ----
        float psum[8];
#pragma unroll
        for (int r = 0; r < 8; ++r) {
            float mx = -1e30f;
#pragma unroll
            for (int t = 0; t < 8; ++t) {
                float s = sca[t][r] * 0.125f;   // 1/sqrt(D)
                sca[t][r] = s;
                mx = fmaxf(mx, s);
            }
            for (int m = 1; m < 16; m <<= 1) mx = fmaxf(mx, __shfl_xor(mx, m, 32));
            const float mup  = fmaxf(rmax[r], mx);
            const float corr = __expf(rmax[r] - mup);
            rmax[r] = mup;
            rsum[r] *= corr;
#pragma unroll
            for (int n = 0; n < 4; ++n) o[n][r] *= corr;
            psum[r] = 0.0f;
        }
#pragma unroll
        for (int t = 0; t < 8; ++t) {
#pragma unroll
            for (int r = 0; r < 8; ++r) {
                const float p = __expf(sca[t][r] - rmax[r]);
                psum[r] += p;
                Ps[wave][r + hi * 8][t * 16 + lh] = (half_t)p;
            }
        }
#pragma unroll
        for (int r = 0; r < 8; ++r) {
            float v = psum[r];
            for (int m = 1; m < 16; m <<= 1) v += __shfl_xor(v, m, 32);
            rsum[r] += v;
        }

        // ---- O += P(16x128) * V(128x64): reload P in A-fragment layout ----
#pragma unroll
        for (int ks = 0; ks < 4; ++ks) {
            const half_t* pp = &Ps[wave][lh][ks * 32 + hi * 8];
            v16h pa = hcat(*(const v8h*)pp, *(const v8h*)(pp + 16));
#pragma unroll
            for (int n = 0; n < 4; ++n) {
                const half_t* vp = &Vs[cur][n * 16 + lh][ks * 32 + hi * 8];
                v16h vf = hcat(*(const v8h*)vp, *(const v8h*)(vp + 16));
                o[n] = wmma_f16(pa, vf, o[n]);
            }
        }
        __syncthreads();  // reads of buf[cur] complete before it is refilled
    }

    // ---- normalize and write ao[b][s][h*64 + d] as f16 ----
    const int b = bh >> 4, h = bh & 15;
#pragma unroll
    for (int r = 0; r < 8; ++r) {
        const float inv = 1.0f / rsum[r];
        const int srow = q0 + r + hi * 8;
        const size_t base = ((size_t)(b * SS + srow)) * EE + h * DD;
#pragma unroll
        for (int n = 0; n < 4; ++n)
            ao[base + n * 16 + lh] = (half_t)(o[n][r] * inv);
    }
}

// ---------------------------------------------------------------------------
// Host-side launch. Workspace layout (bytes):
//  xq16 0, xk16 8M, xv16 16M, wq16 24M, wk16 26M, wv16 28M, wo16 30M,
//  qh 32M, kh 40M, vt 48M, ao 56M   (total 64 MiB)
// ---------------------------------------------------------------------------
extern "C" void kernel_launch(void* const* d_in, const int* in_sizes, int n_in,
                              void* d_out, int out_size, void* d_ws, size_t ws_size,
                              hipStream_t stream) {
    const float* query = (const float*)d_in[0];
    const float* key_  = (const float*)d_in[1];
    const float* value = (const float*)d_in[2];
    const float* Wq = (const float*)d_in[3];
    const float* bq = (const float*)d_in[4];
    const float* Wk = (const float*)d_in[5];
    const float* bk = (const float*)d_in[6];
    const float* Wv = (const float*)d_in[7];
    const float* bv = (const float*)d_in[8];
    const float* Wo = (const float*)d_in[9];
    const float* bo = (const float*)d_in[10];
    float* out = (float*)d_out;

    char* ws = (char*)d_ws;
    const size_t MB = 1u << 20;
    half_t* xq16 = (half_t*)(ws + 0 * MB);
    half_t* xk16 = (half_t*)(ws + 8 * MB);
    half_t* xv16 = (half_t*)(ws + 16 * MB);
    half_t* wq16 = (half_t*)(ws + 24 * MB);
    half_t* wk16 = (half_t*)(ws + 26 * MB);
    half_t* wv16 = (half_t*)(ws + 28 * MB);
    half_t* wo16 = (half_t*)(ws + 30 * MB);
    half_t* qh   = (half_t*)(ws + 32 * MB);
    half_t* kh   = (half_t*)(ws + 40 * MB);
    half_t* vt   = (half_t*)(ws + 48 * MB);
    half_t* ao   = (half_t*)(ws + 56 * MB);

    // 1) convert activations and weights to f16
    const int nAct4 = MTOT * EE / 4;   // 1,048,576
    const int nW4   = EE * EE / 4;     // 262,144
    cvt_f32_f16<<<(nAct4 + 255) / 256, 256, 0, stream>>>(query, xq16, nAct4);
    cvt_f32_f16<<<(nAct4 + 255) / 256, 256, 0, stream>>>(key_,  xk16, nAct4);
    cvt_f32_f16<<<(nAct4 + 255) / 256, 256, 0, stream>>>(value, xv16, nAct4);
    cvt_f32_f16<<<(nW4 + 255) / 256, 256, 0, stream>>>(Wq, wq16, nW4);
    cvt_f32_f16<<<(nW4 + 255) / 256, 256, 0, stream>>>(Wk, wk16, nW4);
    cvt_f32_f16<<<(nW4 + 255) / 256, 256, 0, stream>>>(Wv, wv16, nW4);
    cvt_f32_f16<<<(nW4 + 255) / 256, 256, 0, stream>>>(Wo, wo16, nW4);

    // 2) projections (Linear: x @ W^T + b); W rows are already the Bt layout
    const int gemmBlocks = (MTOT / 128) * (NDIM / 128);  // 256
    gemm_wmma<<<gemmBlocks, 256, 0, stream>>>(xq16, wq16, bq, qh, 0);
    gemm_wmma<<<gemmBlocks, 256, 0, stream>>>(xk16, wk16, bk, kh, 0);
    gemm_wmma<<<gemmBlocks, 256, 0, stream>>>(xv16, wv16, bv, vt, 1);

    // 3) attention: 32 (b,h) x 16 query chunks
    attn_wmma<<<BB * HH * (SS / 128), 256, 0, stream>>>(qh, kh, vt, ao);

    // 4) output projection -> f32 d_out
    gemm_wmma<<<gemmBlocks, 256, 0, stream>>>(ao, wo16, bo, out, 2);
}
